// ScanAssociativeRNNAttention_52398601011220
// MI455X (gfx1250) — matrix operand order repaired and checked by reading, not compiled, not run on hardware
//
#include <hip/hip_runtime.h>
#include <hip/hip_bf16.h>

#define B_   4
#define T_   4096
#define DIN  1024
#define H_   16
#define D_   64
#define NKV  2048          // H_*D_*2 columns of kv GEMM
#define TCH  128           // timesteps per chunk
#define NCH  32            // T_/TCH
#define SUMW 66            // summary record: m, u, w[64]
#define CPAD 130           // LDS row stride for C (conflict padding)
#define SPAD 67            // LDS row stride for scan state
#define OPAD 65            // LDS row stride for output accum

typedef __attribute__((ext_vector_type(16))) __bf16 v16bf;
typedef __attribute__((ext_vector_type(8)))  __bf16 v8bf;
typedef __attribute__((ext_vector_type(8)))  float  v8f;

// ---------------------------------------------------------------------------
// Pre-split kv_kernel (1024 x 2048 fp32, row-major) into transposed bf16
// hi/lo arrays Wt[n][k] so B fragments are two contiguous 16B runs per lane.
// ---------------------------------------------------------------------------
__global__ __launch_bounds__(256) void athena_prep_w(
    const float* __restrict__ kvk, __bf16* __restrict__ Wh, __bf16* __restrict__ Wl)
{
  int idx = blockIdx.x * 256 + threadIdx.x;       // [0, NKV*DIN)
  int n = idx >> 10;
  int k = idx & 1023;
  float x = kvk[(size_t)k * NKV + n];
  __bf16 h = (__bf16)x;
  Wh[(size_t)n * DIN + k] = h;
  Wl[(size_t)n * DIN + k] = (__bf16)(x - (float)h);
}

// ---------------------------------------------------------------------------
// Pre-split inputs (B*T x DIN fp32, row-major) into bf16 hi/lo, same layout.
// Removes all fp32->bf16 VALU work from the WMMA inner loop (it was being
// redone 16 heads x 2 passes = 32x per element).
// ---------------------------------------------------------------------------
__global__ __launch_bounds__(256) void athena_prep_a(
    const float* __restrict__ inp, __bf16* __restrict__ Ah, __bf16* __restrict__ Al)
{
  size_t idx = (size_t)blockIdx.x * 256 + threadIdx.x;   // [0, B_*T_*DIN)
  float x = inp[idx];
  __bf16 h = (__bf16)x;
  Ah[idx] = h;
  Al[idx] = (__bf16)(x - (float)h);
}

// ---------------------------------------------------------------------------
// C[128 x 128] = relu( A[128 x 1024] * W[1024 x 128(head)] ), fp32 accum via
// 3-term bf16 split WMMA (hi*hi + hi*lo + lo*hi). 8 waves: wave = M-tile row,
// each wave covers all 8 N-tiles. Result stored to padded LDS.
// PRE=true: A fragments read directly as bf16 (pure load+WMMA inner loop).
// ---------------------------------------------------------------------------
template <bool PRE>
__device__ __forceinline__ void gemm_head_chunk(
    const float*  __restrict__ A,    // chunk base: [TCH][DIN] fp32 (PRE=false)
    const __bf16* __restrict__ Ah,   // chunk base: [TCH][DIN] bf16 hi (PRE=true)
    const __bf16* __restrict__ Al,   // chunk base: [TCH][DIN] bf16 lo (PRE=true)
    const __bf16* __restrict__ Whd,  // head base of transposed hi: [128][DIN]
    const __bf16* __restrict__ Wld,  // head base of transposed lo
    float* __restrict__ Clds, int tid)
{
  const int wave = tid >> 5;
  const int lane = tid & 31;
  const int lh   = lane & 15;       // A: row-in-tile, B: col-in-tile
  const int kh   = lane >> 4;       // K-half selector

  v8f acc[8] = {};
  const size_t rowoff = (size_t)(wave * 16 + lh) * DIN;

  for (int k0 = 0; k0 < DIN; k0 += 32) {
    // A fragment (16-bit A layout: lanes0-15 K{0..7,16..23}, lanes16-31 +8)
    const int kb = k0 + kh * 8;
    v16bf ahi, alo;
    if constexpr (PRE) {
      union { v16bf v; v8bf h[2]; } uh, ul;
      uh.h[0] = *(const v8bf*)(Ah + rowoff + kb);
      uh.h[1] = *(const v8bf*)(Ah + rowoff + kb + 16);
      ul.h[0] = *(const v8bf*)(Al + rowoff + kb);
      ul.h[1] = *(const v8bf*)(Al + rowoff + kb + 16);
      ahi = uh.v; alo = ul.v;
    } else {
      const float* ap = A + rowoff + kb;
      float4 f0 = *(const float4*)(ap);
      float4 f1 = *(const float4*)(ap + 4);
      float4 f2 = *(const float4*)(ap + 16);
      float4 f3 = *(const float4*)(ap + 20);
      float af[16] = {f0.x, f0.y, f0.z, f0.w, f1.x, f1.y, f1.z, f1.w,
                      f2.x, f2.y, f2.z, f2.w, f3.x, f3.y, f3.z, f3.w};
#pragma unroll
      for (int j = 0; j < 16; ++j) {
        __bf16 h = (__bf16)af[j];
        ahi[j] = h;
        alo[j] = (__bf16)(af[j] - (float)h);
      }
    }
#pragma unroll
    for (int nt = 0; nt < 8; ++nt) {
      size_t boff = (size_t)(nt * 16 + lh) * DIN + kb;
      union { v16bf v; v8bf h[2]; } ubh, ubl;
      ubh.h[0] = *(const v8bf*)(Whd + boff);
      ubh.h[1] = *(const v8bf*)(Whd + boff + 16);
      ubl.h[0] = *(const v8bf*)(Wld + boff);
      ubl.h[1] = *(const v8bf*)(Wld + boff + 16);
      v16bf bh = ubh.v;
      v16bf bl = ubl.v;
      acc[nt] = __builtin_amdgcn_wmma_f32_16x16x32_bf16(false, ahi, false, bh,
                                                        (short)0, acc[nt], false, false);
      acc[nt] = __builtin_amdgcn_wmma_f32_16x16x32_bf16(false, ahi, false, bl,
                                                        (short)0, acc[nt], false, false);
      acc[nt] = __builtin_amdgcn_wmma_f32_16x16x32_bf16(false, alo, false, bh,
                                                        (short)0, acc[nt], false, false);
    }
  }
  // C layout: VGPR r -> M = r + 8*(lane>>4), N = lane&15. Apply relu here.
#pragma unroll
  for (int nt = 0; nt < 8; ++nt) {
#pragma unroll
    for (int r = 0; r < 8; ++r) {
      int m = wave * 16 + r + kh * 8;
      int n = nt * 16 + lh;
      Clds[m * CPAD + n] = fmaxf(acc[nt][r], 0.f);
    }
  }
}

// ---------------------------------------------------------------------------
// Pass 1: per (b, chunk, h) compute chunk summary (M, U, W[64]).
// ---------------------------------------------------------------------------
template <bool PRE>
__global__ __launch_bounds__(256) void athena_summary(
    const float* __restrict__ inputs, const __bf16* __restrict__ Ah,
    const __bf16* __restrict__ Al, const __bf16* __restrict__ Wh,
    const __bf16* __restrict__ Wl, const float* __restrict__ qk,
    float* __restrict__ summ)
{
  extern __shared__ char smem[];
  float* C    = (float*)smem;              // 128*130 f32
  float* sArr = (float*)(smem + 66560);    // 128 f32
  float* eArr = (float*)(smem + 67072);    // 128 f32
  float* qh   = (float*)(smem + 67584);    // 64 f32
  float* Msh  = (float*)(smem + 67840);    // 1 f32

  const int tid   = threadIdx.x;
  const int b     = blockIdx.x / NCH;
  const int chunk = blockIdx.x % NCH;
  const size_t coff = ((size_t)b * T_ + (size_t)chunk * TCH) * DIN;

  for (int h = 0; h < H_; ++h) {
    __syncthreads();
    gemm_head_chunk<PRE>(inputs + coff, PRE ? Ah + coff : nullptr,
                         PRE ? Al + coff : nullptr,
                         Wh + (size_t)h * 128 * DIN, Wl + (size_t)h * 128 * DIN, C, tid);
    if (tid < 64) qh[tid] = qk[h * 64 + tid];
    __syncthreads();
    if (tid < TCH) {                       // s[t] = q . k[t]
      float s = 0.f;
      const float* cr = C + tid * CPAD;
#pragma unroll 8
      for (int d = 0; d < D_; ++d) s += qh[d] * cr[2 * d];
      sArr[tid] = s;
    }
    __syncthreads();
    if (tid == 0) {
      float M = -__builtin_inff();
      for (int t = 0; t < TCH; ++t) M = fmaxf(M, sArr[t]);
      Msh[0] = M;
    }
    __syncthreads();
    if (tid < TCH) eArr[tid] = __expf(sArr[tid] - Msh[0]);
    __syncthreads();
    float* sp = summ + (((size_t)b * H_ + h) * NCH + chunk) * SUMW;
    if (tid == 0) {
      float U = 0.f;
      for (int t = 0; t < TCH; ++t) U += eArr[t];
      sp[0] = Msh[0]; sp[1] = U;
    }
    if (tid < 64) {
      float W = 0.f;
      for (int t = 0; t < TCH; ++t) W += eArr[t] * C[t * CPAD + 2 * tid + 1];
      sp[2 + tid] = W;
    }
  }
}

// ---------------------------------------------------------------------------
// Pass 2: per (b, chunk): for each head, combine cross-chunk prefix,
// recompute chunk GEMM, Hillis-Steele scan over 128 t in LDS, accumulate
// w/u over heads, store (b, t, d) slab.
// ---------------------------------------------------------------------------
template <bool PRE>
__global__ __launch_bounds__(256) void athena_final(
    const float* __restrict__ inputs, const __bf16* __restrict__ Ah,
    const __bf16* __restrict__ Al, const __bf16* __restrict__ Wh,
    const __bf16* __restrict__ Wl, const float* __restrict__ qk,
    const float* __restrict__ summ, float* __restrict__ out)
{
  extern __shared__ char smem[];
  float* C    = (float*)smem;               // 128*130 f32 (reused as scan buf B)
  float* Sa   = (float*)(smem + 66560);     // 128*67 f32 (scan buf A)
  float* Oacc = (float*)(smem + 100864);    // 128*65 f32
  float* sArr = (float*)(smem + 134144);    // 128 f32
  float* qh   = (float*)(smem + 134656);    // 64 f32

  const int tid   = threadIdx.x;
  const int b     = blockIdx.x / NCH;
  const int chunk = blockIdx.x % NCH;
  const size_t coff = ((size_t)b * T_ + (size_t)chunk * TCH) * DIN;

  for (int i = tid; i < TCH * OPAD; i += 256) Oacc[i] = 0.f;

  for (int h = 0; h < H_; ++h) {
    // exclusive prefix over chunks 0..chunk-1 (registers; lanes 0..63 own w[d])
    float pm = -__builtin_inff(), pu = 0.f, pw = 0.f;
    if (tid < 64) {
      for (int cc = 0; cc < chunk; ++cc) {
        const float* sp = summ + (((size_t)b * H_ + h) * NCH + cc) * SUMW;
        float mc = sp[0], uc = sp[1], wc = sp[2 + tid];
        float m  = fmaxf(pm, mc);
        float ea = __expf(pm - m), eb = __expf(mc - m);
        pu = pu * ea + uc * eb;
        pw = pw * ea + wc * eb;
        pm = m;
      }
    }
    __syncthreads();
    gemm_head_chunk<PRE>(inputs + coff, PRE ? Ah + coff : nullptr,
                         PRE ? Al + coff : nullptr,
                         Wh + (size_t)h * 128 * DIN, Wl + (size_t)h * 128 * DIN, C, tid);
    if (tid < 64) qh[tid] = qk[h * 64 + tid];
    __syncthreads();
    if (tid < TCH) {
      float s = 0.f;
      const float* cr = C + tid * CPAD;
#pragma unroll 8
      for (int d = 0; d < D_; ++d) s += qh[d] * cr[2 * d];
      sArr[tid] = s;
    }
    __syncthreads();
    {   // build scan elems: Sa[t] = { s_t, 1, v_t[0..63] }
      int t = tid >> 1, part = tid & 1;
      float* sa = Sa + t * SPAD;
      const float* cr = C + t * CPAD;
      if (part == 0) { sa[0] = sArr[t]; sa[1] = 1.f; }
      for (int d = part * 32; d < part * 32 + 32; ++d) sa[2 + d] = cr[2 * d + 1];
    }
    __syncthreads();
    if (tid < 64) {   // fold prefix into element 0 (identity when chunk==0)
      float s0 = sArr[0];
      float m0 = fmaxf(pm, s0);
      float ea = __expf(pm - m0), eb = __expf(s0 - m0);
      float v0 = Sa[2 + tid];
      Sa[2 + tid] = pw * ea + v0 * eb;
      if (tid == 0) { Sa[0] = m0; Sa[1] = pu * ea + eb; }
    }
    __syncthreads();
    // Hillis-Steele inclusive scan, 7 passes, ping-pong Sa <-> C
    float* src = Sa;
    float* dst = C;
    for (int off = 1; off < TCH; off <<= 1) {
      int t = tid >> 1, part = tid & 1;
      float* d_ = dst + t * SPAD;
      const float* b_ = src + t * SPAD;
      if (t >= off) {
        const float* a_ = src + (t - off) * SPAD;
        float ma = a_[0], mb = b_[0];
        float m  = fmaxf(ma, mb);
        float ea = __expf(ma - m), eb = __expf(mb - m);
        if (part == 0) {
          d_[0] = m;
          d_[1] = a_[1] * ea + b_[1] * eb;
          for (int f = 2; f < 33; ++f) d_[f] = a_[f] * ea + b_[f] * eb;
        } else {
          for (int f = 33; f < 66; ++f) d_[f] = a_[f] * ea + b_[f] * eb;
        }
      } else {
        if (part == 0) { for (int f = 0;  f < 33; ++f) d_[f] = b_[f]; }
        else           { for (int f = 33; f < 66; ++f) d_[f] = b_[f]; }
      }
      __syncthreads();
      float* tmp = src; src = dst; dst = tmp;
    }
    {   // h = w/u, accumulate over heads
      int t = tid >> 1, part = tid & 1;
      const float* r = src + t * SPAD;
      float inv = 1.f / r[1];
      float* oa = Oacc + t * OPAD;
      for (int d = part * 32; d < part * 32 + 32; ++d) oa[d] += r[2 + d] * inv;
    }
    __syncthreads();
  }
  for (int i = tid; i < TCH * D_; i += 256) {
    int t = i >> 6, d = i & 63;
    out[((size_t)b * T_ + (size_t)chunk * TCH + t) * D_ + d] = Oacc[t * OPAD + d];
  }
}

// ---------------------------------------------------------------------------
extern "C" void kernel_launch(void* const* d_in, const int* in_sizes, int n_in,
                              void* d_out, int out_size, void* d_ws, size_t ws_size,
                              hipStream_t stream) {
  (void)in_sizes; (void)n_in; (void)out_size;
  const float* inputs = (const float*)d_in[0];
  const float* kvk    = (const float*)d_in[1];
  const float* qk     = (const float*)d_in[2];
  float* out = (float*)d_out;

  char* ws = (char*)d_ws;
  __bf16* Wh = (__bf16*)ws;                                  // 4 MB
  __bf16* Wl = (__bf16*)(ws + (size_t)4  * 1024 * 1024);     // 4 MB
  float* summ = (float*)(ws + (size_t)8  * 1024 * 1024);     // 540 KB
  __bf16* Ah = (__bf16*)(ws + (size_t)16 * 1024 * 1024);     // 32 MB
  __bf16* Al = (__bf16*)(ws + (size_t)48 * 1024 * 1024);     // 32 MB
  const bool pre = ws_size >= (size_t)80 * 1024 * 1024;

  const int lds1 = 68096;    // summary kernel LDS bytes
  const int lds2 = 135168;   // final kernel LDS bytes

  athena_prep_w<<<(NKV * DIN) / 256, 256, 0, stream>>>(kvk, Wh, Wl);
  if (pre) {
    (void)hipFuncSetAttribute(reinterpret_cast<const void*>(&athena_summary<true>),
                              hipFuncAttributeMaxDynamicSharedMemorySize, lds1);
    (void)hipFuncSetAttribute(reinterpret_cast<const void*>(&athena_final<true>),
                              hipFuncAttributeMaxDynamicSharedMemorySize, lds2);
    athena_prep_a<<<(int)(((size_t)B_ * T_ * DIN) / 256), 256, 0, stream>>>(inputs, Ah, Al);
    athena_summary<true><<<B_ * NCH, 256, lds1, stream>>>(inputs, Ah, Al, Wh, Wl, qk, summ);
    athena_final<true><<<B_ * NCH, 256, lds2, stream>>>(inputs, Ah, Al, Wh, Wl, qk, summ, out);
  } else {
    (void)hipFuncSetAttribute(reinterpret_cast<const void*>(&athena_summary<false>),
                              hipFuncAttributeMaxDynamicSharedMemorySize, lds1);
    (void)hipFuncSetAttribute(reinterpret_cast<const void*>(&athena_final<false>),
                              hipFuncAttributeMaxDynamicSharedMemorySize, lds2);
    athena_summary<false><<<B_ * NCH, 256, lds1, stream>>>(inputs, nullptr, nullptr, Wh, Wl, qk, summ);
    athena_final<false><<<B_ * NCH, 256, lds2, stream>>>(inputs, nullptr, nullptr, Wh, Wl, qk, summ, out);
  }
}
